// MultiheadAttention_7438883357444
// MI455X (gfx1250) — matrix operand order repaired and checked by reading, not compiled, not run on hardware
//
#include <hip/hip_runtime.h>
#include <hip/hip_bf16.h>

#define D_MODEL 1024
#define NHEADS  16
#define HEADDIM 64
#define BATCH   8
#define SEQ     1024

typedef __bf16 bf16_t;
typedef __attribute__((ext_vector_type(16))) __bf16 v16bf;
typedef __attribute__((ext_vector_type(8)))  float  v8f;
typedef __attribute__((ext_vector_type(4)))  unsigned int u32x4;

union FragU { v16bf v; u32x4 q[2]; };

#define ZERO8 ((v8f){0.f,0.f,0.f,0.f,0.f,0.f,0.f,0.f})

__device__ __forceinline__ v8f wmma_bf16(v16bf a, v16bf b, v8f c) {
  // D = A(16x32) * B(32x16) + C, fp32 accumulate
  return __builtin_amdgcn_wmma_f32_16x16x32_bf16(
      /*neg_a=*/false, a, /*neg_b=*/false, b,
      /*c_mod=*/(short)0, c, /*reuse_a=*/false, /*reuse_b=*/false);
}

// A-fragment: 16x32 tile of row-major [*, ld] bf16 matrix at (m0, k0).
// ISA layout: lane(0-15)=row M, half=lane>>4; elem e -> K = k0 + 8*half + e (e<8)
//                                               K = k0 + 16 + 8*half + (e-8) (e>=8)
__device__ __forceinline__ v16bf load_afrag(const bf16_t* base, int ld, int m0, int k0) {
  const int lane = threadIdx.x & 31;
  const int hlf  = lane >> 4;
  const int m    = lane & 15;
  const bf16_t* p = base + (size_t)(m0 + m) * ld + (k0 + 8 * hlf);
  FragU f;
  f.q[0] = *(const u32x4*)(p);
  f.q[1] = *(const u32x4*)(p + 16);
  return f.v;
}

// B-fragment: 32x16 tile of B at (k0, n0), where BT (= B column-major) is given
// row-major [N][ld]. ISA layout: lane%16 = col N; elem e -> K = k0 + 8*half + e (e<8), k0 + 16 + 8*half + (e-8) (e>=8).
__device__ __forceinline__ v16bf load_bfrag(const bf16_t* baseT, int ld, int k0, int n0) {
  const int lane = threadIdx.x & 31;
  const int hlf  = lane >> 4;
  const int n    = lane & 15;
  const bf16_t* p = baseT + (size_t)(n0 + n) * ld + (k0 + 8 * hlf);
  FragU f;
  f.q[0] = *(const u32x4*)(p);
  f.q[1] = *(const u32x4*)(p + 16);
  return f.v;
}

// ---------------- precision-convert kernels ----------------
__global__ void cvt_bf16_kernel(const float* __restrict__ src, bf16_t* __restrict__ dst, int n) {
  for (int i = blockIdx.x * blockDim.x + threadIdx.x; i < n; i += gridDim.x * blockDim.x)
    dst[i] = (bf16_t)src[i];
}

// src [K][N] fp32 -> dst [N][K] bf16 (weights stored K-major for B-fragments)
__global__ void cvt_transpose_bf16_kernel(const float* __restrict__ src, bf16_t* __restrict__ dst,
                                          int K, int N) {
  const int total = K * N;
  for (int i = blockIdx.x * blockDim.x + threadIdx.x; i < total; i += gridDim.x * blockDim.x) {
    const int k = i / N;
    const int n = i - k * N;
    dst[(size_t)n * K + k] = (bf16_t)src[i];
  }
}

// ---------------- fused QKV projection ----------------
// qkv[8192,3072] = x[8192,1024] @ Wqkv + bqkv ; scatter to Q,K [B,H,S,Hd], V^T [B,H,Hd,S]
__global__ void __launch_bounds__(256) qkv_gemm_kernel(
    const bf16_t* __restrict__ xb, const bf16_t* __restrict__ WqkvT,
    const float* __restrict__ bqkv,
    bf16_t* __restrict__ Qb, bf16_t* __restrict__ Kb, bf16_t* __restrict__ VTb) {
  const int lane = threadIdx.x & 31;
  const int hlf  = lane >> 4;
  const int lm   = lane & 15;
  const int wave = threadIdx.x >> 5;
  const int m_wave = blockIdx.y * 128 + (wave >> 1) * 32;   // 4 waves along M
  const int n_wave = blockIdx.x * 128 + (wave & 1) * 64;    // 2 waves along N

  v8f acc[2][4];
#pragma unroll
  for (int i = 0; i < 2; ++i)
#pragma unroll
    for (int j = 0; j < 4; ++j) acc[i][j] = ZERO8;

  for (int k0 = 0; k0 < D_MODEL; k0 += 64) {
    if (k0 + 64 < D_MODEL) {
      __builtin_prefetch(xb + (size_t)m_wave * D_MODEL + k0 + 64, 0, 3);
      __builtin_prefetch(WqkvT + (size_t)n_wave * D_MODEL + k0 + 64, 0, 3);
    }
    // batch all 24 b128 loads into distinct locals so the scheduler can keep
    // them in flight under the WMMA chain (avoid s_wait_loadcnt 0 lockstep)
    v16bf a[2][2];
    a[0][0] = load_afrag(xb, D_MODEL, m_wave,      k0);
    a[0][1] = load_afrag(xb, D_MODEL, m_wave,      k0 + 32);
    a[1][0] = load_afrag(xb, D_MODEL, m_wave + 16, k0);
    a[1][1] = load_afrag(xb, D_MODEL, m_wave + 16, k0 + 32);
    v16bf bf[4][2];
#pragma unroll
    for (int j = 0; j < 4; ++j) {
      bf[j][0] = load_bfrag(WqkvT, D_MODEL, k0,      n_wave + 16 * j);
      bf[j][1] = load_bfrag(WqkvT, D_MODEL, k0 + 32, n_wave + 16 * j);
    }
#pragma unroll
    for (int j = 0; j < 4; ++j) {
      acc[0][j] = wmma_bf16(a[0][0], bf[j][0], acc[0][j]);
      acc[0][j] = wmma_bf16(a[0][1], bf[j][1], acc[0][j]);
      acc[1][j] = wmma_bf16(a[1][0], bf[j][0], acc[1][j]);
      acc[1][j] = wmma_bf16(a[1][1], bf[j][1], acc[1][j]);
    }
  }

  // epilogue: bias + scatter (C/D layout: lane%16 = col, vgpr r -> row r + 8*half)
#pragma unroll
  for (int i = 0; i < 2; ++i) {
#pragma unroll
    for (int j = 0; j < 4; ++j) {
      const int n    = n_wave + 16 * j + lm;
      const float bias = bqkv[n];
      const int part = n >> 10;        // 0=q 1=k 2=v
      const int rem  = n & 1023;
      const int h    = rem >> 6;
      const int hd   = rem & 63;
#pragma unroll
      for (int r = 0; r < 8; ++r) {
        const int t  = m_wave + 16 * i + r + 8 * hlf;   // flat token
        const int bb = t >> 10;
        const int s  = t & 1023;
        const bf16_t val = (bf16_t)(acc[i][j][r] + bias);
        if (part == 0)
          Qb[(((size_t)(bb * NHEADS + h)) * SEQ + s) * HEADDIM + hd] = val;
        else if (part == 1)
          Kb[(((size_t)(bb * NHEADS + h)) * SEQ + s) * HEADDIM + hd] = val;
        else
          VTb[(((size_t)(bb * NHEADS + h)) * HEADDIM + hd) * SEQ + s] = val;
      }
    }
  }
}

// ---------------- flash attention ----------------
// grid (S/64, B*H), 4 waves/block; wave w owns queries [q0, q0+16)
__global__ void __launch_bounds__(128) attn_kernel(
    const bf16_t* __restrict__ Qb, const bf16_t* __restrict__ Kb,
    const bf16_t* __restrict__ VTb, bf16_t* __restrict__ vals) {
  __shared__ __align__(16) bf16_t pbuf[4][16 * 64];   // per-wave P tile

  const int lane = threadIdx.x & 31;
  const int hlf  = lane >> 4;
  const int lm   = lane & 15;
  const int wave = threadIdx.x >> 5;
  const int bh   = blockIdx.y;
  const int b    = bh >> 4;
  const int h    = bh & 15;
  const int q0   = blockIdx.x * 64 + wave * 16;

  const bf16_t* Qh = Qb  + ((size_t)bh * SEQ) * HEADDIM;   // [S][64]
  const bf16_t* Kh = Kb  + ((size_t)bh * SEQ) * HEADDIM;   // [S][64]
  const bf16_t* Vt = VTb + ((size_t)bh * HEADDIM) * SEQ;   // [64][S]

  // Q fragments resident for the whole pass (Hd = 64 = 2 k-chunks)
  const v16bf qf0 = load_afrag(Qh, HEADDIM, q0, 0);
  const v16bf qf1 = load_afrag(Qh, HEADDIM, q0, 32);

  float mstat[8], lstat[8];
  v8f acc[4];
#pragma unroll
  for (int r = 0; r < 8; ++r) { mstat[r] = -3.0e38f; lstat[r] = 0.f; }
#pragma unroll
  for (int j = 0; j < 4; ++j) acc[j] = ZERO8;

  for (int kb = 0; kb < SEQ; kb += 64) {
    // ---- preload K fragments (K stored [key][hd] == (K^T) column-major) ----
    v16bf bk[4][2];
#pragma unroll
    for (int j = 0; j < 4; ++j) {
      bk[j][0] = load_bfrag(Kh, HEADDIM, 0,  kb + 16 * j);
      bk[j][1] = load_bfrag(Kh, HEADDIM, 32, kb + 16 * j);
    }
    // ---- preload V fragments NOW: softmax VALU/LDS work hides their latency ----
    v16bf bv[4][2];
#pragma unroll
    for (int j = 0; j < 4; ++j) {
      bv[j][0] = load_bfrag(Vt, SEQ, kb,      16 * j);
      bv[j][1] = load_bfrag(Vt, SEQ, kb + 32, 16 * j);
    }
    // ---- scores: S = Q @ K^T ----
    v8f sc[4];
#pragma unroll
    for (int j = 0; j < 4; ++j) {
      v8f z = ZERO8;
      z = wmma_bf16(qf0, bk[j][0], z);
      z = wmma_bf16(qf1, bk[j][1], z);
      sc[j] = z;
    }
    // ---- online softmax (row M=r+8*half lives in one 16-lane half) ----
    float alpha[8];
#pragma unroll
    for (int r = 0; r < 8; ++r) {
      float rmax = -3.0e38f;
#pragma unroll
      for (int j = 0; j < 4; ++j) {
        sc[j][r] *= 0.125f;                      // 1/sqrt(64)
        rmax = fmaxf(rmax, sc[j][r]);
      }
#pragma unroll
      for (int off = 8; off >= 1; off >>= 1)
        rmax = fmaxf(rmax, __shfl_xor(rmax, off, 32));
      const float newm = fmaxf(mstat[r], rmax);
      alpha[r] = __expf(mstat[r] - newm);
      mstat[r] = newm;
      float rsum = 0.f;
#pragma unroll
      for (int j = 0; j < 4; ++j) {
        const float p = __expf(sc[j][r] - newm);
        rsum += p;
        pbuf[wave][(r + 8 * hlf) * 64 + 16 * j + lm] = (bf16_t)p;
      }
#pragma unroll
      for (int off = 8; off >= 1; off >>= 1)
        rsum += __shfl_xor(rsum, off, 32);
      lstat[r] = lstat[r] * alpha[r] + rsum;
    }
    __syncthreads();
    // ---- P: C-layout -> A-fragment layout via LDS ----
    v16bf pa0, pa1;
    {
      const bf16_t* pb = &pbuf[wave][lm * 64 + 8 * hlf];
      FragU f;
      f.q[0] = *(const u32x4*)(pb);
      f.q[1] = *(const u32x4*)(pb + 16);
      pa0 = f.v;
      f.q[0] = *(const u32x4*)(pb + 32);
      f.q[1] = *(const u32x4*)(pb + 48);
      pa1 = f.v;
    }
    // ---- O = O*alpha + P @ V (V^T stored [hd][S] == V column-major) ----
#pragma unroll
    for (int j = 0; j < 4; ++j) {
#pragma unroll
      for (int r = 0; r < 8; ++r) acc[j][r] *= alpha[r];
      acc[j] = wmma_bf16(pa0, bv[j][0], acc[j]);
      acc[j] = wmma_bf16(pa1, bv[j][1], acc[j]);
    }
    __syncthreads();
  }

  // epilogue: normalize, write vals [B,S,D] bf16 for output projection
#pragma unroll
  for (int j = 0; j < 4; ++j) {
    const int col = h * HEADDIM + 16 * j + lm;
#pragma unroll
    for (int r = 0; r < 8; ++r) {
      const int q = q0 + r + 8 * hlf;
      const float o = acc[j][r] / lstat[r];
      vals[((size_t)(b * SEQ + q)) * D_MODEL + col] = (bf16_t)o;
    }
  }
}

// ---------------- output projection ----------------
__global__ void __launch_bounds__(256) out_gemm_kernel(
    const bf16_t* __restrict__ vals, const bf16_t* __restrict__ WoT,
    const float* __restrict__ bo, float* __restrict__ out) {
  const int lane = threadIdx.x & 31;
  const int hlf  = lane >> 4;
  const int lm   = lane & 15;
  const int wave = threadIdx.x >> 5;
  const int m_wave = blockIdx.y * 128 + (wave >> 1) * 32;
  const int n_wave = blockIdx.x * 128 + (wave & 1) * 64;

  v8f acc[2][4];
#pragma unroll
  for (int i = 0; i < 2; ++i)
#pragma unroll
    for (int j = 0; j < 4; ++j) acc[i][j] = ZERO8;

  for (int k0 = 0; k0 < D_MODEL; k0 += 64) {
    if (k0 + 64 < D_MODEL) {
      __builtin_prefetch(vals + (size_t)m_wave * D_MODEL + k0 + 64, 0, 3);
      __builtin_prefetch(WoT + (size_t)n_wave * D_MODEL + k0 + 64, 0, 3);
    }
    v16bf a[2][2];
    a[0][0] = load_afrag(vals, D_MODEL, m_wave,      k0);
    a[0][1] = load_afrag(vals, D_MODEL, m_wave,      k0 + 32);
    a[1][0] = load_afrag(vals, D_MODEL, m_wave + 16, k0);
    a[1][1] = load_afrag(vals, D_MODEL, m_wave + 16, k0 + 32);
    v16bf bw[4][2];
#pragma unroll
    for (int j = 0; j < 4; ++j) {
      bw[j][0] = load_bfrag(WoT, D_MODEL, k0,      n_wave + 16 * j);
      bw[j][1] = load_bfrag(WoT, D_MODEL, k0 + 32, n_wave + 16 * j);
    }
#pragma unroll
    for (int j = 0; j < 4; ++j) {
      acc[0][j] = wmma_bf16(a[0][0], bw[j][0], acc[0][j]);
      acc[0][j] = wmma_bf16(a[0][1], bw[j][1], acc[0][j]);
      acc[1][j] = wmma_bf16(a[1][0], bw[j][0], acc[1][j]);
      acc[1][j] = wmma_bf16(a[1][1], bw[j][1], acc[1][j]);
    }
  }
#pragma unroll
  for (int i = 0; i < 2; ++i)
#pragma unroll
    for (int j = 0; j < 4; ++j) {
      const int n = n_wave + 16 * j + lm;
      const float bias = bo[n];
#pragma unroll
      for (int r = 0; r < 8; ++r) {
        const int t = m_wave + 16 * i + r + 8 * hlf;
        out[(size_t)t * D_MODEL + n] = acc[i][j][r] + bias;
      }
    }
}

extern "C" void kernel_launch(void* const* d_in, const int* in_sizes, int n_in,
                              void* d_out, int out_size, void* d_ws, size_t ws_size,
                              hipStream_t stream) {
  (void)in_sizes; (void)n_in; (void)out_size; (void)ws_size;
  const float* x    = (const float*)d_in[0];
  const float* Wqkv = (const float*)d_in[1];
  const float* bqkv = (const float*)d_in[2];
  const float* Wo   = (const float*)d_in[3];
  const float* bo   = (const float*)d_in[4];
  float* out = (float*)d_out;

  char* ws = (char*)d_ws;
  size_t off = 0;
  auto walloc = [&](size_t bytes) -> void* {
    void* p = ws + off;
    off += (bytes + 255) & ~(size_t)255;
    return p;
  };
  const size_t NTOK = (size_t)BATCH * SEQ;  // 8192
  bf16_t* xb    = (bf16_t*)walloc(NTOK * D_MODEL * sizeof(bf16_t));                // 16 MB
  bf16_t* WqkvT = (bf16_t*)walloc((size_t)3 * D_MODEL * D_MODEL * sizeof(bf16_t)); // 6 MB
  bf16_t* WoT   = (bf16_t*)walloc((size_t)D_MODEL * D_MODEL * sizeof(bf16_t));     // 2 MB
  bf16_t* Qb    = (bf16_t*)walloc(NTOK * D_MODEL * sizeof(bf16_t));                // 16 MB
  bf16_t* Kb    = (bf16_t*)walloc(NTOK * D_MODEL * sizeof(bf16_t));                // 16 MB
  bf16_t* VTb   = (bf16_t*)walloc(NTOK * D_MODEL * sizeof(bf16_t));                // 16 MB
  bf16_t* vals  = (bf16_t*)walloc(NTOK * D_MODEL * sizeof(bf16_t));                // 16 MB

  cvt_bf16_kernel<<<4096, 256, 0, stream>>>(x, xb, (int)(NTOK * D_MODEL));
  cvt_transpose_bf16_kernel<<<4096, 256, 0, stream>>>(Wqkv, WqkvT, D_MODEL, 3 * D_MODEL);
  cvt_transpose_bf16_kernel<<<2048, 256, 0, stream>>>(Wo, WoT, D_MODEL, D_MODEL);
  qkv_gemm_kernel<<<dim3(24, 64), 256, 0, stream>>>(xb, WqkvT, bqkv, Qb, Kb, VTb);
  attn_kernel<<<dim3(16, 128), 128, 0, stream>>>(Qb, Kb, VTb, vals);
  out_gemm_kernel<<<dim3(8, 64), 256, 0, stream>>>(vals, WoT, bo, out);
}
